// GAT_11201274708231
// MI455X (gfx1250) — compile-verified, matrix operand order checked by reading, not run on hardware
//
#include <hip/hip_runtime.h>
#include <hip/hip_bf16.h>

#define NEG_SLOPE 0.2f

typedef __attribute__((ext_vector_type(16))) __bf16 v16bf;
typedef __attribute__((ext_vector_type(8)))  float  v8f;

// -------- order-preserving float <-> uint key (for atomic segment-max) --------
__device__ __forceinline__ unsigned fkey(float f) {
  unsigned u = __float_as_uint(f);
  return (u & 0x80000000u) ? ~u : (u | 0x80000000u);
}
__device__ __forceinline__ float funkey(unsigned k) {
  unsigned u = (k & 0x80000000u) ? (k & 0x7fffffffu) : ~k;
  return __uint_as_float(u);
}

// ------------------------------- small utilities ------------------------------
__global__ void k_zero_u32(unsigned* __restrict__ p, size_t n) {
  size_t i = (size_t)blockIdx.x * blockDim.x + threadIdx.x;
  if (i < n) p[i] = 0u;
}

__global__ void k_cvt_bf16(const float* __restrict__ in, __bf16* __restrict__ out, size_t n) {
  size_t i = (size_t)blockIdx.x * blockDim.x + threadIdx.x;
  if (i < n) out[i] = (__bf16)in[i];
}

// in: [K, N] row-major fp32  ->  out: [N, K] row-major bf16 (B pre-transposed for WMMA)
__global__ void k_transpose_bf16(const float* __restrict__ in, __bf16* __restrict__ out,
                                 int K, int N) {
  int i = blockIdx.x * blockDim.x + threadIdx.x;
  if (i >= K * N) return;
  int k = i / N, n = i % N;
  out[(size_t)n * K + k] = (__bf16)in[i];
}

// ------------------------------ WMMA bf16 GEMM --------------------------------
// C[M,N] (fp32) = A[M,K] (bf16 row-major) * Bt[N,K]^T (bf16, pre-transposed).
// One wave per 16x16 tile; K stepped by 32 via v_wmma_f32_16x16x32_bf16.
// 16-bit A-fragment lane layout (ISA 7.12.2): lane<16 -> row=lane, half=0;
// lane>=16 -> row=lane-16, half=1; elements 0..7 <-> k = 8*half+0..7,
// elements 8..15 <-> k = 16+8*half+0..7 (consecutive-K pairs per dword).
__global__ __launch_bounds__(256) void k_gemm_bf16_wmma(
    const __bf16* __restrict__ A, const __bf16* __restrict__ Bt,
    float* __restrict__ C, int M, int K, int N) {
  int wave   = blockIdx.x * (blockDim.x >> 5) + (threadIdx.x >> 5);
  int ntiles = N >> 4;
  int mtiles = M >> 4;
  if (wave >= mtiles * ntiles) return;           // wave-uniform: EXEC stays all-1s
  int mt = wave / ntiles, nt = wave % ntiles;
  int lane = threadIdx.x & 31;
  int half = lane >> 4;
  int r    = lane & 15;

  const __bf16* arow = A  + (size_t)(mt * 16 + r) * K + half * 8;
  const __bf16* brow = Bt + (size_t)(nt * 16 + r) * K + half * 8;

  v8f acc = {};
  for (int k0 = 0; k0 < K; k0 += 32) {
    union { v16bf v; unsigned u[8]; } fa, fb;
    const unsigned* pa0 = (const unsigned*)(arow + k0);
    const unsigned* pa1 = (const unsigned*)(arow + k0 + 16);
    const unsigned* pb0 = (const unsigned*)(brow + k0);
    const unsigned* pb1 = (const unsigned*)(brow + k0 + 16);
#pragma unroll
    for (int j = 0; j < 4; ++j) {
      fa.u[j] = pa0[j]; fa.u[4 + j] = pa1[j];
      fb.u[j] = pb0[j]; fb.u[4 + j] = pb1[j];
    }
    acc = __builtin_amdgcn_wmma_f32_16x16x32_bf16(
        /*neg_a=*/false, fa.v, /*neg_b=*/false, fb.v,
        /*c_mod=*/(short)0, acc, /*reuse_a=*/false, /*reuse_b=*/false);
  }

  // D layout: VGPR q -> row = 16*mt + 8*half + q, col = 16*nt + (lane&15)
  float* out = C + (size_t)(mt * 16 + half * 8) * N + nt * 16 + r;
#pragma unroll
  for (int q = 0; q < 8; ++q) out[(size_t)q * N] = acc[q];
}

// ---------------------- per-node attention coefficients -----------------------
__global__ void k_att(const float* __restrict__ xl, const float* __restrict__ att_s,
                      const float* __restrict__ att_d, float* __restrict__ a_s,
                      float* __restrict__ a_d, int Nn, int H, int C) {
  int i = blockIdx.x * blockDim.x + threadIdx.x;     // over Nn*H
  if (i >= Nn * H) return;
  int h = i % H;
  int n = i / H;
  const float* row = xl + (size_t)n * H * C + (size_t)h * C;
  float ss = 0.f, sd = 0.f;
  for (int c = 0; c < C; ++c) {
    float v = row[c];
    ss += v * att_s[h * C + c];
    sd += v * att_d[h * C + c];
  }
  a_s[i] = ss;
  a_d[i] = sd;
}

// ------------------------- segment max over dst (edges) -----------------------
__global__ void k_edge_max(const int* __restrict__ src, const int* __restrict__ dst,
                           const float* __restrict__ a_s, const float* __restrict__ a_d,
                           unsigned* __restrict__ mkey, int E, int Nn, int H) {
  long long idx   = (long long)blockIdx.x * blockDim.x + threadIdx.x;
  long long total = (long long)(E + Nn) * H;
  if (idx >= total) return;
  int h = (int)(idx % H);
  long long e = idx / H;
  int s, d;
  if (e < E) { s = src[e]; d = dst[e]; } else { s = d = (int)(e - E); }  // self-loops
  float al = a_s[(size_t)s * H + h] + a_d[(size_t)d * H + h];
  al = al > 0.f ? al : al * NEG_SLOPE;
  atomicMax(&mkey[(size_t)d * H + h], fkey(al));
}

// -------- fused exp / denominator / weighted-message scatter (unnormalized) ---
__global__ void k_edge_scatter(const int* __restrict__ src, const int* __restrict__ dst,
                               const float* __restrict__ a_s, const float* __restrict__ a_d,
                               const unsigned* __restrict__ mkey, const float* __restrict__ xl,
                               float* __restrict__ accum, float* __restrict__ den,
                               int E, int Nn, int H, int C) {
  size_t idx   = (size_t)blockIdx.x * blockDim.x + threadIdx.x;
  size_t total = (size_t)(E + Nn) * (size_t)(H * C);
  if (idx >= total) return;
  int    HC = H * C;
  int    hc = (int)(idx % HC);
  size_t e  = idx / HC;
  int    h  = hc / C;
  int s, d;
  if (e < (size_t)E) { s = src[e]; d = dst[e]; } else { s = d = (int)(e - E); }
  float al = a_s[(size_t)s * H + h] + a_d[(size_t)d * H + h];
  al = al > 0.f ? al : al * NEG_SLOPE;
  float ex = __expf(al - funkey(mkey[(size_t)d * H + h]));
  if ((hc & (C - 1)) == 0)                       // one den add per (edge, head); C=32
    atomicAdd(&den[(size_t)d * H + h], ex);
  atomicAdd(&accum[(size_t)d * HC + hc], xl[(size_t)s * HC + hc] * ex);
}

// ------------- layer-1 epilogue: normalize + bias + ELU -> bf16 h -------------
__global__ void k_final1(const float* __restrict__ accum, const float* __restrict__ den,
                         const float* __restrict__ bias, __bf16* __restrict__ hout,
                         int Nn, int H, int C) {
  size_t idx   = (size_t)blockIdx.x * blockDim.x + threadIdx.x;
  size_t total = (size_t)Nn * H * C;
  if (idx >= total) return;
  int    HC = H * C;
  int    hc = (int)(idx % HC);
  size_t n  = idx / HC;
  int    h  = hc / C;
  float v = accum[idx] / (den[n * H + h] + 1e-16f) + bias[hc];
  v = v > 0.f ? v : (__expf(v) - 1.f);           // ELU(alpha=1)
  hout[idx] = (__bf16)v;
}

// -------- layer-2 epilogue: normalize per head, mean over heads, + bias -------
__global__ void k_final2(const float* __restrict__ accum, const float* __restrict__ den,
                         const float* __restrict__ bias, float* __restrict__ out,
                         int Nn, int H, int C) {
  int idx = blockIdx.x * blockDim.x + threadIdx.x;   // over Nn*C
  if (idx >= Nn * C) return;
  int n = idx / C, c = idx % C;
  float s = 0.f;
#pragma unroll
  for (int h = 0; h < 8; ++h)
    s += accum[(size_t)n * H * C + h * C + c] / (den[(size_t)n * H + h] + 1e-16f);
  out[idx] = s * (1.f / (float)H) + bias[c];
}

// =============================== host launcher ================================
extern "C" void kernel_launch(void* const* d_in, const int* in_sizes, int n_in,
                              void* d_out, int out_size, void* d_ws, size_t ws_size,
                              hipStream_t stream) {
  (void)n_in; (void)out_size; (void)ws_size;
  const float* x      = (const float*)d_in[0];
  const int*   ei     = (const int*)d_in[1];
  const float* W1     = (const float*)d_in[2];
  const float* att_s1 = (const float*)d_in[3];
  const float* att_d1 = (const float*)d_in[4];
  const float* bias1  = (const float*)d_in[5];
  const float* W2     = (const float*)d_in[6];
  const float* att_s2 = (const float*)d_in[7];
  const float* att_d2 = (const float*)d_in[8];
  const float* bias2  = (const float*)d_in[9];

  const int IN_CH = 256, H1 = 4, H2 = 8, C = 32;
  const int Nn = in_sizes[0] / IN_CH;   // 50000
  const int E  = in_sizes[1] / 2;       // 800000
  const int F1 = H1 * C;                // 128
  const int F2 = H2 * C;                // 256
  const int* src = ei;
  const int* dst = ei + E;

  // ---------------- workspace carve-up (256B aligned) ----------------
  char*  base = (char*)d_ws;
  size_t off  = 0;
  auto take = [&](size_t bytes) -> void* {
    void* p = base + off;
    off = (off + bytes + 255) & ~(size_t)255;
    return p;
  };
  __bf16*   x_bf   = (__bf16*)take((size_t)Nn * IN_CH * 2);
  __bf16*   W1t    = (__bf16*)take((size_t)F1 * IN_CH * 2);   // [F1, IN_CH]
  __bf16*   W2t    = (__bf16*)take((size_t)F2 * F1 * 2);      // [F2, F1]
  __bf16*   h_bf   = (__bf16*)take((size_t)Nn * F1 * 2);
  float*    xl1    = (float*) take((size_t)Nn * F2 * 4);      // region reused as xl2
  float*    accum1 = xl1 + (size_t)Nn * F1;
  float*    xl2    = xl1;
  float*    accum2 = (float*) take((size_t)Nn * F2 * 4);
  float*    a_s1   = (float*) take((size_t)Nn * H1 * 4);
  float*    a_d1   = (float*) take((size_t)Nn * H1 * 4);
  unsigned* m1     = (unsigned*)take((size_t)Nn * H1 * 4);
  float*    den1   = (float*) take((size_t)Nn * H1 * 4);
  float*    a_s2   = (float*) take((size_t)Nn * H2 * 4);
  float*    a_d2   = (float*) take((size_t)Nn * H2 * 4);
  unsigned* m2     = (unsigned*)take((size_t)Nn * H2 * 4);
  float*    den2   = (float*) take((size_t)Nn * H2 * 4);

  const int T = 256;
  auto nb = [](size_t n, int t) { return (unsigned)((n + (size_t)t - 1) / t); };

  // ---------------- precision prep + zero-init ----------------
  k_cvt_bf16<<<nb((size_t)Nn * IN_CH, T), T, 0, stream>>>(x, x_bf, (size_t)Nn * IN_CH);
  k_transpose_bf16<<<nb((size_t)IN_CH * F1, T), T, 0, stream>>>(W1, W1t, IN_CH, F1);
  k_transpose_bf16<<<nb((size_t)F1 * F2, T), T, 0, stream>>>(W2, W2t, F1, F2);
  k_zero_u32<<<nb((size_t)Nn * F1, T), T, 0, stream>>>((unsigned*)accum1, (size_t)Nn * F1);
  k_zero_u32<<<nb((size_t)Nn * F2, T), T, 0, stream>>>((unsigned*)accum2, (size_t)Nn * F2);
  k_zero_u32<<<nb((size_t)Nn * H1, T), T, 0, stream>>>(m1, (size_t)Nn * H1);      // key(0) < key(any finite)
  k_zero_u32<<<nb((size_t)Nn * H1, T), T, 0, stream>>>((unsigned*)den1, (size_t)Nn * H1);
  k_zero_u32<<<nb((size_t)Nn * H2, T), T, 0, stream>>>(m2, (size_t)Nn * H2);
  k_zero_u32<<<nb((size_t)Nn * H2, T), T, 0, stream>>>((unsigned*)den2, (size_t)Nn * H2);

  // ---------------- layer 1 ----------------
  {
    int tiles = (Nn / 16) * (F1 / 16);   // 3125 * 8
    k_gemm_bf16_wmma<<<(tiles + 7) / 8, 256, 0, stream>>>(x_bf, W1t, xl1, Nn, IN_CH, F1);
  }
  k_att<<<nb((size_t)Nn * H1, T), T, 0, stream>>>(xl1, att_s1, att_d1, a_s1, a_d1, Nn, H1, C);
  k_edge_max<<<nb((size_t)(E + Nn) * H1, T), T, 0, stream>>>(src, dst, a_s1, a_d1, m1, E, Nn, H1);
  k_edge_scatter<<<nb((size_t)(E + Nn) * F1, T), T, 0, stream>>>(
      src, dst, a_s1, a_d1, m1, xl1, accum1, den1, E, Nn, H1, C);
  k_final1<<<nb((size_t)Nn * F1, T), T, 0, stream>>>(accum1, den1, bias1, h_bf, Nn, H1, C);

  // ---------------- layer 2 ----------------
  {
    int tiles = (Nn / 16) * (F2 / 16);   // 3125 * 16
    k_gemm_bf16_wmma<<<(tiles + 7) / 8, 256, 0, stream>>>(h_bf, W2t, xl2, Nn, F1, F2);
  }
  k_att<<<nb((size_t)Nn * H2, T), T, 0, stream>>>(xl2, att_s2, att_d2, a_s2, a_d2, Nn, H2, C);
  k_edge_max<<<nb((size_t)(E + Nn) * H2, T), T, 0, stream>>>(src, dst, a_s2, a_d2, m2, E, Nn, H2);
  k_edge_scatter<<<nb((size_t)(E + Nn) * F2, T), T, 0, stream>>>(
      src, dst, a_s2, a_d2, m2, xl2, accum2, den2, E, Nn, H2, C);
  k_final2<<<nb((size_t)Nn * C, T), T, 0, stream>>>(accum2, den2, bias2, (float*)d_out, Nn, H2, C);
}
